// NNUE_16381005267418
// MI455X (gfx1250) — matrix operand order repaired and checked by reading, not compiled, not run on hardware
//
#include <hip/hip_runtime.h>
#include <stdint.h>

#define FEAT 4476
#define NACT 30
#define BATCH 16384
#define H 512
#define MTILE 64          // batch rows per block
#define NWAVES 16         // 512 threads / block
#define NTASK (2 * MTILE) // (row, side) feature-transform tasks per block

typedef __attribute__((ext_vector_type(16))) _Float16 v16h;
typedef __attribute__((ext_vector_type(8)))  float    v8f;
typedef __attribute__((ext_vector_type(8)))  int      v8i;

union Frag { v8i i; v16h h; };

// LDS partition (bytes): idx | a0 (64x1024 f16) | a1 (64x512) | a2 (64x256) | a3 (64x128)
#define OFF_IDX 0
#define OFF_A0  (NTASK * NACT * 4)                 // 15360
#define OFF_A1  (OFF_A0 + MTILE * 1024 * 2)        // +131072
#define OFF_A2  (OFF_A1 + MTILE * 512 * 2)         // +65536
#define OFF_A3  (OFF_A2 + MTILE * 256 * 2)         // +32768
#define SMEM_BYTES (OFF_A3 + MTILE * 128 * 2)      // 261120 <= 320KB WGP LDS

// ---- prep: transpose ft_w [H][F] -> [F][H] (f32, contiguous per feature) ----
__global__ void k_transpose_ftw(const float* __restrict__ ftw, float* __restrict__ out) {
  int t = blockIdx.x * 256 + threadIdx.x;
  if (t >= FEAT * H) return;
  int h = t & (H - 1);
  int f = t >> 9;
  out[t] = ftw[(size_t)h * FEAT + f];
}

// ---- prep: f32 -> f16 weight conversion (layout preserved [N][K]) ----
__global__ void k_f32_to_f16(const float* __restrict__ src, _Float16* __restrict__ dst, int n) {
  int t = blockIdx.x * 256 + threadIdx.x;
  if (t < n) dst[t] = (_Float16)src[t];
}

// 64xN GEMM layer with bias+ReLU. A in LDS [64][K] f16, W in global [N][K] f16.
// Each wave owns N-tiles (stride NWAVES*16) and sweeps 4 M-tiles per B fragment,
// so every weight fragment is loaded once and fed into 4 WMMAs.
__device__ __forceinline__ void gemm_relu_64xN(
    const _Float16* s_in, _Float16* s_out,
    const _Float16* __restrict__ wgt, const float* __restrict__ bias,
    int K, int N, int lane, int wave)
{
  const int m0 = lane & 15;       // A-row-within-tile / B-col for this lane
  const int hh = lane >> 4;       // lane half selects K sub-block
  for (int nt = wave * 16; nt < N; nt += NWAVES * 16) {
    v8f acc[4];
#pragma unroll
    for (int mt = 0; mt < 4; ++mt)
      acc[mt] = (v8f){0.f, 0.f, 0.f, 0.f, 0.f, 0.f, 0.f, 0.f};
    const uint32_t* wrow = (const uint32_t*)(wgt + (size_t)(nt + m0) * K);
    for (int kb = 0; kb < K; kb += 32) {
      const int kd = kb >> 1;     // dword offset of this K block
      Frag b;
#pragma unroll
      for (int v = 0; v < 8; ++v)  // B 32x16 f16 layout: K = hh*16 + 2v, N = m0
        b.i[v] = wrow[kd + (hh << 3) + v];
#pragma unroll
      for (int mt = 0; mt < 4; ++mt) {
        Frag a;
        const uint32_t* arow = (const uint32_t*)(s_in + (mt * 16 + m0) * K);
#pragma unroll
        for (int v = 0; v < 8; ++v)  // A 16x32 f16 layout: K = hh*16 + 2v
          a.i[v] = arow[kd + (hh << 3) + v];
        acc[mt] = __builtin_amdgcn_wmma_f32_16x16x32_f16(false, a.h, false, b.h,
                                                         (short)0, acc[mt], false, false);
      }
    }
    const int n = nt + m0;
    const float bv = bias[n];
#pragma unroll
    for (int mt = 0; mt < 4; ++mt) {
#pragma unroll
      for (int r = 0; r < 8; ++r) {  // D row = mt*16 + hh*8 + r, col = n
        float x = acc[mt][r] + bv;
        x = x > 0.f ? x : 0.f;
        s_out[(mt * 16 + hh * 8 + r) * N + n] = (_Float16)x;
      }
    }
  }
}

__global__ __launch_bounds__(NWAVES * 32) void k_nnue(
    const int* __restrict__ wi, const int* __restrict__ bi,
    const float* __restrict__ ftw_t,
    const _Float16* __restrict__ w1h, const float* __restrict__ b1,
    const _Float16* __restrict__ w2h, const float* __restrict__ b2,
    const _Float16* __restrict__ w3h, const float* __restrict__ b3,
    const float* __restrict__ wo, const float* __restrict__ bo,
    float* __restrict__ out)
{
  extern __shared__ __align__(16) char smem[];
  int*      s_idx = (int*)(smem + OFF_IDX);        // [NTASK][NACT]
  _Float16* s_a0  = (_Float16*)(smem + OFF_A0);    // [64][1024] concat FT output
  _Float16* s_a1  = (_Float16*)(smem + OFF_A1);    // [64][512]
  _Float16* s_a2  = (_Float16*)(smem + OFF_A2);    // [64][256]
  _Float16* s_a3  = (_Float16*)(smem + OFF_A3);    // [64][128]

  const int tid  = threadIdx.x;
  const int lane = tid & 31;
  const int wave = tid >> 5;
  const int b0   = blockIdx.x * MTILE;

  // ---- stage 0: stage the 128 (row,side) index lists into LDS ----
  for (int t = tid; t < NTASK * NACT; t += NWAVES * 32) {
    int task = t / NACT, j = t % NACT;
    int row = task & (MTILE - 1), side = task >> 6;
    s_idx[task * NACT + j] = side ? bi[(b0 + row) * NACT + j]
                                  : wi[(b0 + row) * NACT + j];
  }
  __syncthreads();

  // ---- stage 1: sparse feature transform, one wave per (row,side) task ----
  for (int task = wave; task < NTASK; task += NWAVES) {
    const int row = task & (MTILE - 1), side = task >> 6;
    float acc[16];
#pragma unroll
    for (int e = 0; e < 16; ++e) acc[e] = 0.f;
    for (int j = 0; j < NACT; ++j) {
      const int id = s_idx[task * NACT + j];
      bool dup = false;                 // one-hot SET semantics: skip duplicates
      for (int k = 0; k < j; ++k) dup = dup || (s_idx[task * NACT + k] == id);
      if (!dup) {                       // uniform across wave -> scalar branch
        const float* col = ftw_t + (size_t)id * H + lane * 16;
#pragma unroll
        for (int e = 0; e < 16; ++e) acc[e] += col[e];
      }
    }
#pragma unroll
    for (int e = 0; e < 16; ++e) {
      float x = acc[e];
      x = x > 1.f ? 1.f : (x < -1.f ? -1.f : x);
      s_a0[row * 1024 + side * 512 + lane * 16 + e] = (_Float16)x;
    }
  }
  __syncthreads();

  // ---- stages 2-4: fused dense ReLU layers via WMMA ----
  gemm_relu_64xN(s_a0, s_a1, w1h, b1, 1024, 512, lane, wave);
  __syncthreads();
  gemm_relu_64xN(s_a1, s_a2, w2h, b2, 512, 256, lane, wave);
  __syncthreads();
  gemm_relu_64xN(s_a2, s_a3, w3h, b3, 256, 128, lane, wave);
  __syncthreads();

  // ---- stage 5: final 128-dot per row ----
  if (tid < MTILE) {
    float s = bo[0];
    const _Float16* a3 = s_a3 + tid * 128;
#pragma unroll 8
    for (int k = 0; k < 128; ++k) s += (float)a3[k] * wo[k];
    out[b0 + tid] = s;
  }
}

extern "C" void kernel_launch(void* const* d_in, const int* in_sizes, int n_in,
                              void* d_out, int out_size, void* d_ws, size_t ws_size,
                              hipStream_t stream) {
  const int*   wi  = (const int*)d_in[0];
  const int*   bi  = (const int*)d_in[1];
  const float* ftw = (const float*)d_in[2];
  const float* w1  = (const float*)d_in[3];
  const float* b1  = (const float*)d_in[4];
  const float* w2  = (const float*)d_in[5];
  const float* b2  = (const float*)d_in[6];
  const float* w3  = (const float*)d_in[7];
  const float* b3  = (const float*)d_in[8];
  const float* wo  = (const float*)d_in[9];
  const float* bo  = (const float*)d_in[10];
  float* out = (float*)d_out;

  // workspace: ft_w^T (f32) + f16 copies of w1/w2/w3  (~10.5 MB total)
  char* ws = (char*)d_ws;
  size_t off = 0;
  float*    ftw_t = (float*)(ws + off); off += (size_t)FEAT * H * 4; off = (off + 255) & ~(size_t)255;
  _Float16* w1h   = (_Float16*)(ws + off); off += (size_t)512 * 1024 * 2;
  _Float16* w2h   = (_Float16*)(ws + off); off += (size_t)256 * 512  * 2;
  _Float16* w3h   = (_Float16*)(ws + off); off += (size_t)128 * 256  * 2;

  const int n_ftw = FEAT * H;
  k_transpose_ftw<<<(n_ftw + 255) / 256, 256, 0, stream>>>(ftw, ftw_t);
  k_f32_to_f16<<<(512 * 1024 + 255) / 256, 256, 0, stream>>>(w1, w1h, 512 * 1024);
  k_f32_to_f16<<<(256 * 512  + 255) / 256, 256, 0, stream>>>(w2, w2h, 256 * 512);
  k_f32_to_f16<<<(128 * 256  + 255) / 256, 256, 0, stream>>>(w3, w3h, 128 * 256);

  k_nnue<<<BATCH / MTILE, NWAVES * 32, SMEM_BYTES, stream>>>(
      wi, bi, ftw_t, w1h, b1, w2h, b2, w3h, b3, wo, bo, out);
}
